// NEP_53755810677167
// MI455X (gfx1250) — compile-verified, matrix operand order checked by reading, not run on hardware
//
#include <hip/hip_runtime.h>
#include <hip/hip_bf16.h>
#include <hip/hip_fp16.h>

// ---------------------------------------------------------------------------
// NEP descriptor + MLP for MI455X (gfx1250, wave32, WMMA).
//   Kernel 1: per-atom descriptor. VALU for chebyshev/blm, WMMA f32 16x16x4
//             for s = gna^T (5x224) * blm (224x24). Staging buffers are
//             K-major (transposed) so each WMMA operand pair is one
//             ds_load_b64 at an immediate offset; dead lanes read dedicated
//             zero rows, keeping all strides uniform (no cndmask/exec masks).
//   Kernel 2: per-16-atom-tile MLP with V_WMMA_F32_16X16X32_F16 (both types
//             computed, per-atom type select).
//   Kernel 3: deterministic tree reduction Ei -> Etot.
// ---------------------------------------------------------------------------

typedef __attribute__((ext_vector_type(16))) _Float16 v16h;
typedef __attribute__((ext_vector_type(8)))  float    v8f;
typedef __attribute__((ext_vector_type(2)))  float    v2f;

#ifndef __has_builtin
#define __has_builtin(x) 0
#endif
#if __has_builtin(__builtin_amdgcn_wmma_f32_16x16x4_f32)
#define HAVE_WMMA_F32X4 1
#else
#define HAVE_WMMA_F32X4 0
#endif

// ---- problem constants (match reference) ----
#define NTYPES_  2
#define MAXN_    100
#define MM_      200      // M = MAXN*NTYPES
#define MPAD_    224      // M padded to multiple of 32 (and of K=4)
#define GSTR_    226      // K-major row stride (even for b64 align; 226%64=34
                          //  -> 16 rows map to 16 distinct LDS banks)
#define NBASE_   13       // NBASER+1 == NBASEA+1
#define NGN_     5        // NMAXR+1 == NMAXA+1
#define FEAT_    35
#define KPAD_    64       // feature K padded for f16 WMMA (2 steps of 32)
#define NEURON_  100
#define NPAD_    112      // neurons padded to 7*16
#define RC_R_    8.0f
#define RC_A_    4.0f
#define PI_F     3.14159265358979323846f

__constant__ float C3Bc[24] = {
  0.238732414637843f, 0.119366207318922f, 0.119366207318922f,
  0.099471839432435f, 0.596831036594608f, 0.596831036594608f,
  0.149207759148652f, 0.149207759148652f, 0.139260575205408f,
  0.104445431404056f, 0.104445431404056f, 1.044454314040563f,
  1.044454314040563f, 0.174075719006761f, 0.174075719006761f,
  0.011190581936149f, 0.223811638722978f, 0.223811638722978f,
  0.111905819361489f, 0.111905819361489f, 1.566681471060845f,
  1.566681471060845f, 0.195835183882606f, 0.195835183882606f };
__constant__ float C4Bc[5] = {
  -0.007499480826664f, -0.134990654879954f, 0.067495327439977f,
   0.404971964639861f, -0.809943929279723f };
__constant__ float C5Bc[3] = {
  0.026596810706114f, 0.053193621412227f, 0.026596810706114f };

// g_n(r) = sum_k c[n][k] * 0.5*(T_k(x)+1)*fc ; accumulates into out[5].
__device__ __forceinline__ void cheb_gn(float r, float rcInv, float fc,
                                        const float* __restrict__ c,
                                        float out[NGN_]) {
  float tt = r * rcInv - 1.0f;
  float xx = 2.0f * tt * tt - 1.0f;
  float Tm2 = 1.0f, Tm1 = xx;
  float f = fc;                        // k = 0: 0.5*(1+1)*fc
  #pragma unroll
  for (int n = 0; n < NGN_; ++n) out[n] += c[n * NBASE_ + 0] * f;
  f = 0.5f * (xx + 1.0f) * fc;         // k = 1
  #pragma unroll
  for (int n = 0; n < NGN_; ++n) out[n] += c[n * NBASE_ + 1] * f;
  #pragma unroll
  for (int k = 2; k < NBASE_; ++k) {
    float T = 2.0f * xx * Tm1 - Tm2; Tm2 = Tm1; Tm1 = T;
    f = 0.5f * (T + 1.0f) * fc;
    #pragma unroll
    for (int n = 0; n < NGN_; ++n) out[n] += c[n * NBASE_ + k] * f;
  }
}

// ---------------------------------------------------------------------------
// Kernel 1: one wave (32 threads) per atom.
// ---------------------------------------------------------------------------
__global__ __launch_bounds__(32)
void nep_descriptor_kernel(const float* __restrict__ ImageDR,
                           const int*   __restrict__ tmap,
                           const float* __restrict__ c2,
                           const float* __restrict__ c3,
                           _Float16*    __restrict__ featsWS,
                           int NAT) {
  __shared__ float c2L[NTYPES_ * NGN_ * NBASE_];   // [tj][n][k], 130 f32
  __shared__ float c3L[NTYPES_ * NGN_ * NBASE_];
  // K-major staging: row = matrix row (n or c), column = neighbor m.
  // One extra zero row each feeds WMMA lanes whose row/col is padding.
  __shared__ float gnaT[(NGN_ + 1) * GSTR_];       // rows 0..4 = gna, row 5 = 0
  __shared__ float blmT[25 * GSTR_];               // rows 0..23 = blm, row 24 = 0
  __shared__ float sL[NGN_ * 24];                  // s[n][c]

  const int lane = threadIdx.x;
  const int g    = blockIdx.x;          // flattened atom b*NAT + i
  const int i    = g % NAT;
  const int ti   = tmap[i];

  for (int idx = lane; idx < NTYPES_ * NGN_ * NBASE_; idx += 32) {
    int tj  = idx / (NGN_ * NBASE_);
    int rem = idx % (NGN_ * NBASE_);
    c2L[idx] = c2[(ti * NTYPES_ + tj) * (NGN_ * NBASE_) + rem];
    c3L[idx] = c3[(ti * NTYPES_ + tj) * (NGN_ * NBASE_) + rem];
  }
  for (int idx = lane; idx < GSTR_; idx += 32) {   // zero rows for dead lanes
    gnaT[NGN_ * GSTR_ + idx] = 0.f;
    blmT[24 * GSTR_ + idx]   = 0.f;
  }
  __syncthreads();

  float q2a[NGN_] = {0.f, 0.f, 0.f, 0.f, 0.f};
  const size_t base = (size_t)g * MM_ * 4;

  // ---- phase 1: per-neighbor VALU work, stage gna/blm (K-major) ----
  for (int m = lane; m < MPAD_; m += 32) {
    float gn[NGN_] = {0.f, 0.f, 0.f, 0.f, 0.f};
    float bl[24];
    #pragma unroll
    for (int c = 0; c < 24; ++c) bl[c] = 0.f;

    if (m < MM_) {
      if (m + 32 < MM_)
        __builtin_prefetch(ImageDR + base + (size_t)(m + 32) * 4, 0, 0);
      float4 dr = *reinterpret_cast<const float4*>(ImageDR + base + (size_t)m * 4);
      float r = dr.x;
      const int tj = (m >= MAXN_) ? 1 : 0;
      const float* cr = c2L + tj * (NGN_ * NBASE_);
      const float* ca = c3L + tj * (NGN_ * NBASE_);

      bool vR = (r > 1e-5f) && (r < RC_R_);
      bool vA = (r > 1e-5f) && (r < RC_A_);
      float fcr = vR ? 0.5f * (__cosf(PI_F * r * (1.0f / RC_R_)) + 1.0f) : 0.0f;
      float fca = vA ? 0.5f * (__cosf(PI_F * r * (1.0f / RC_A_)) + 1.0f) : 0.0f;

      cheb_gn(r, 1.0f / RC_R_, fcr, cr, q2a);  // radial channel -> q2 accum
      cheb_gn(r, 1.0f / RC_A_, fca, ca, gn);   // angular channel -> gna

      float rinv = 1.0f / fmaxf(r, 1e-12f);
      float x = dr.y * rinv, y = dr.z * rinv, z = dr.w * rinv;
      float x2 = x * x, y2 = y * y, z2 = z * z;
      float xy2 = x2 - y2;
      bl[0] = z;  bl[1] = x;  bl[2] = y;
      bl[3] = 3.f * z2 - 1.f;           bl[4] = x * z;  bl[5] = y * z;
      bl[6] = xy2;                      bl[7] = 2.f * x * y;
      bl[8]  = (5.f * z2 - 3.f) * z;    bl[9]  = (5.f * z2 - 1.f) * x;
      bl[10] = (5.f * z2 - 1.f) * y;    bl[11] = xy2 * z;
      bl[12] = 2.f * x * y * z;         bl[13] = (x2 - 3.f * y2) * x;
      bl[14] = (3.f * x2 - y2) * y;
      bl[15] = (35.f * z2 - 30.f) * z2 + 3.f;
      bl[16] = (7.f * z2 - 3.f) * x * z; bl[17] = (7.f * z2 - 3.f) * y * z;
      bl[18] = (7.f * z2 - 1.f) * xy2;   bl[19] = (7.f * z2 - 1.f) * 2.f * x * y;
      bl[20] = (x2 - 3.f * y2) * x * z;  bl[21] = (3.f * x2 - y2) * y * z;
      bl[22] = xy2 * xy2 - 4.f * x2 * y2; bl[23] = 4.f * x * y * xy2;
    }

    // K-major stores: consecutive lanes hit consecutive addresses (no
    // bank conflicts); pad neighbors (m >= MM_) store zeros.
    #pragma unroll
    for (int n = 0; n < NGN_; ++n) gnaT[n * GSTR_ + m] = gn[n];
    #pragma unroll
    for (int c = 0; c < 24; ++c)   blmT[c * GSTR_ + m] = bl[c];
  }
  __syncthreads();

  // ---- phase 2: s[n][c] = sum_m gna[m][n] * blm[m][c] ----
#if HAVE_WMMA_F32X4
  {
    // A = gna^T : 16(rows n, 5 used) x 4(K=m).  B = blm : 4 x 16 columns c.
    // Every lane reads its K-pair as one b64 load; padding lanes read the
    // zero rows so all streams advance uniformly (+4 elems per K-step) and
    // all loop offsets are immediates.
    const int rowN = lane & 15;
    const int col  = lane & 15;
    const int kb   = (lane >> 4) * 2;     // K lane striping {0,1}/{2,3}

    const float* ga  = gnaT + ((rowN < NGN_) ? rowN : NGN_) * GSTR_ + kb;
    const float* bb0 = blmT + col * GSTR_ + kb;
    const float* bb1 = blmT + ((col < 8) ? (16 + col) : 24) * GSTR_ + kb;

    v8f s0 = {}; v8f s1 = {};
    for (int kk = 0; kk < MPAD_; kk += 4) {
      float2 af  = *reinterpret_cast<const float2*>(ga + kk);
      float2 b0f = *reinterpret_cast<const float2*>(bb0 + kk);
      float2 b1f = *reinterpret_cast<const float2*>(bb1 + kk);
      v2f a, b0, b1v;
      a.x = af.x;   a.y = af.y;
      b0.x = b0f.x; b0.y = b0f.y;
      b1v.x = b1f.x; b1v.y = b1f.y;
      s0 = __builtin_amdgcn_wmma_f32_16x16x4_f32(false, a, false, b0,
                                                 (short)0, s0, false, false);
      s1 = __builtin_amdgcn_wmma_f32_16x16x4_f32(false, a, false, b1v,
                                                 (short)0, s1, false, false);
    }
    // D layout: lanes 0-15 hold rows M=0..7 in vgprs 0..7 -> rows n=0..4.
    if (lane < 16) {
      #pragma unroll
      for (int rr = 0; rr < NGN_; ++rr) sL[rr * 24 + lane] = s0[rr];
      if (lane < 8) {
        #pragma unroll
        for (int rr = 0; rr < NGN_; ++rr) sL[rr * 24 + 16 + lane] = s1[rr];
      }
    }
  }
#else
  for (int idx = lane; idx < NGN_ * 24; idx += 32) {
    int n = idx / 24, c = idx % 24;
    float acc = 0.f;
    for (int m = 0; m < MM_; ++m) acc += gnaT[n * GSTR_ + m] * blmT[c * GSTR_ + m];
    sL[idx] = acc;
  }
#endif
  __syncthreads();

  // ---- phase 3: wave-reduce q2, build 35 features, emit f16 padded ----
  #pragma unroll
  for (int n = 0; n < NGN_; ++n) {
    float v = q2a[n];
    #pragma unroll
    for (int off = 16; off; off >>= 1) v += __shfl_xor(v, off, 32);
    q2a[n] = v;
  }

  _Float16* fw = featsWS + (size_t)g * KPAD_;
  if (lane < NGN_) {
    const int n = lane;
    const float* sn = &sL[n * 24];
    fw[n] = (_Float16)q2a[n];
    const int offs[5] = {0, 3, 8, 15, 24};
    #pragma unroll
    for (int l = 0; l < 4; ++l) {
      float acc = 0.f;
      for (int c = offs[l]; c < offs[l + 1]; ++c) {
        float v = sn[c]; acc += C3Bc[c] * v * v;
      }
      fw[NGN_ + n * 4 + l] = (_Float16)acc;
    }
    float s3 = sn[3], s4 = sn[4], s5 = sn[5], s6 = sn[6], s7 = sn[7];
    float q4 = C4Bc[0] * s3 * s3 * s3
             + C4Bc[1] * s3 * (s4 * s4 + s5 * s5)
             + C4Bc[2] * s3 * (s6 * s6 + s7 * s7)
             + C4Bc[3] * s6 * (s5 * s5 - s4 * s4)
             + C4Bc[4] * s4 * s5 * s7;
    fw[25 + n] = (_Float16)q4;
    float s0s = sn[0] * sn[0];
    float s12 = sn[1] * sn[1] + sn[2] * sn[2];
    float q5 = C5Bc[0] * s0s * s0s + C5Bc[1] * s0s * s12 + C5Bc[2] * s12 * s12;
    fw[30 + n] = (_Float16)q5;
  }
  for (int idx = FEAT_ + lane; idx < KPAD_; idx += 32) fw[idx] = (_Float16)0.f;
}

// ---------------------------------------------------------------------------
// Kernel 2: MLP. 128 threads = 4 waves; each wave owns a 16-atom tile.
// f16 WMMA fragments follow ISA 7.12.2 16-bit layouts:
//   A: lane row = lane&15; VGPR j holds K = (j>>2)*16 + (lane>>4)*8 + (j&3)*2 {+0,+1}
//   B: lane col = lane&15; same K striping (mirrored layout).
// ---------------------------------------------------------------------------
__global__ __launch_bounds__(128)
void nep_mlp_kernel(const _Float16* __restrict__ featsWS,
                    const int*      __restrict__ tmap,
                    const float*    __restrict__ W1,
                    const float*    __restrict__ b1,
                    const float*    __restrict__ W2,
                    const float*    __restrict__ b2,
                    float*          __restrict__ Ei,
                    int NAT) {
  __shared__ _Float16 W1h[NTYPES_ * NPAD_ * KPAD_];   // [t][n][k] f16, zero-pad

  const int tid = threadIdx.x;
  for (int idx = tid; idx < NTYPES_ * NPAD_ * KPAD_; idx += 128) {
    int t   = idx / (NPAD_ * KPAD_);
    int rem = idx % (NPAD_ * KPAD_);
    int n   = rem / KPAD_;
    int k   = rem % KPAD_;
    float v = (k < FEAT_ && n < NEURON_) ? W1[(t * FEAT_ + k) * NEURON_ + n] : 0.f;
    W1h[idx] = (_Float16)v;
  }
  __syncthreads();

  const int lane = tid & 31;
  const int wv   = tid >> 5;
  const int tileBase = (blockIdx.x * 4 + wv) * 16;
  const int row = lane & 15;
  const int hb  = lane >> 4;
  const int colBase = lane & 15;

  union Frag { v16h v; unsigned u[8]; };
  Frag A0, A1;
  const _Float16* fr = featsWS + (size_t)(tileBase + row) * KPAD_;
  #pragma unroll
  for (int j = 0; j < 8; ++j) {
    int K = ((j >> 2) << 4) + (hb << 3) + ((j & 3) << 1);
    A0.u[j] = *reinterpret_cast<const unsigned*>(fr + K);       // K-step 0
    A1.u[j] = *reinterpret_cast<const unsigned*>(fr + 32 + K);  // K-step 1
  }

  for (int t = 0; t < NTYPES_; ++t) {
    v8f acc[7] = {};
    #pragma unroll
    for (int nt = 0; nt < 7; ++nt) {
      int n = nt * 16 + colBase;
      Frag B0, B1;
      const _Float16* wp = &W1h[(t * NPAD_ + n) * KPAD_];
      #pragma unroll
      for (int j = 0; j < 8; ++j) {
        int K = ((j >> 2) << 4) + (hb << 3) + ((j & 3) << 1);
        B0.u[j] = *reinterpret_cast<const unsigned*>(wp + K);
        B1.u[j] = *reinterpret_cast<const unsigned*>(wp + 32 + K);
      }
      acc[nt] = __builtin_amdgcn_wmma_f32_16x16x32_f16(false, A0.v, false, B0.v,
                                                       (short)0, acc[nt], false, false);
      acc[nt] = __builtin_amdgcn_wmma_f32_16x16x32_f16(false, A1.v, false, B1.v,
                                                       (short)0, acc[nt], false, false);
    }

    float b1v[7], w2v[7];
    #pragma unroll
    for (int nt = 0; nt < 7; ++nt) {
      int c = nt * 16 + colBase;
      bool ok = c < NEURON_;
      b1v[nt] = ok ? b1[t * NEURON_ + c] : 0.f;
      w2v[nt] = ok ? W2[t * NEURON_ + c] : 0.f;
    }
    float bb2 = b2[t];

    #pragma unroll
    for (int r = 0; r < 8; ++r) {
      float p = 0.f;
      #pragma unroll
      for (int nt = 0; nt < 7; ++nt)
        p += tanhf(acc[nt][r] + b1v[nt]) * w2v[nt];
      #pragma unroll
      for (int off = 8; off; off >>= 1) p += __shfl_xor(p, off, 16);
      if ((lane & 15) == 0) {
        int aIdx = tileBase + r + 8 * hb;   // D row = r + 8*(lane>>4)
        int ii = aIdx % NAT;
        if (tmap[ii] == t) Ei[aIdx] = p + bb2;
      }
    }
  }
}

// ---------------------------------------------------------------------------
// Kernel 3: Etot[b] = sum_i Ei[b][i] (fixed-order tree reduction).
// ---------------------------------------------------------------------------
__global__ __launch_bounds__(256)
void nep_etot_kernel(const float* __restrict__ Ei, float* __restrict__ Etot,
                     int NAT) {
  __shared__ float red[256];
  const int b = blockIdx.x, tid = threadIdx.x;
  float s = 0.f;
  for (int i = tid; i < NAT; i += 256) s += Ei[(size_t)b * NAT + i];
  red[tid] = s;
  __syncthreads();
  for (int off = 128; off; off >>= 1) {
    if (tid < off) red[tid] += red[tid + off];
    __syncthreads();
  }
  if (tid == 0) Etot[b] = red[0];
}

// ---------------------------------------------------------------------------
extern "C" void kernel_launch(void* const* d_in, const int* in_sizes, int n_in,
                              void* d_out, int out_size, void* d_ws, size_t ws_size,
                              hipStream_t stream) {
  const float* ImageDR = (const float*)d_in[0];
  const int*   tmap    = (const int*)  d_in[1];
  const float* c2      = (const float*)d_in[2];
  const float* c3      = (const float*)d_in[3];
  const float* W1      = (const float*)d_in[4];
  const float* b1      = (const float*)d_in[5];
  const float* W2      = (const float*)d_in[6];
  const float* b2      = (const float*)d_in[7];
  float* out = (float*)d_out;

  const int NAT = in_sizes[1];                 // Imagetype_map length
  const int B   = out_size / (NAT + 1);        // out = Etot[B] ++ Ei[B*NAT]
  const int nAtoms = B * NAT;

  _Float16* featsWS = (_Float16*)d_ws;         // nAtoms * KPAD_ * 2 bytes

  nep_descriptor_kernel<<<nAtoms, 32, 0, stream>>>(ImageDR, tmap, c2, c3,
                                                   featsWS, NAT);
  const int tiles = nAtoms / 16;
  nep_mlp_kernel<<<tiles / 4, 128, 0, stream>>>(featsWS, tmap, W1, b1, W2, b2,
                                                out + B, NAT);
  nep_etot_kernel<<<B, 256, 0, stream>>>(out + B, out, NAT);
}
